// SS2D_9234179686983
// MI455X (gfx1250) — matrix operand order repaired and checked by reading, not compiled, not run on hardware
//
#include <hip/hip_runtime.h>
#include <hip/hip_bf16.h>
#include <math.h>

// ---------------- problem constants ----------------
#define Bn   4
#define Cc   64
#define Hh   128
#define Ww   128
#define Ll   (Hh*Ww)      // 16384
#define Dd   128
#define Ns   8
#define Rr   4
#define DBLc (Rr + 2*Ns)  // 20
#define NCH  64           // scan chunks
#define CHL  (Ll/NCH)     // 256

typedef _Float16 f16;
typedef __attribute__((ext_vector_type(16))) _Float16 v16h;
typedef __attribute__((ext_vector_type(8)))  float    v8f;
typedef __attribute__((ext_vector_type(4)))  float    fv4;
typedef __attribute__((ext_vector_type(2)))  float    fv2;

union Frag { v16h v; fv4 f4[2]; };
union H2   { f16 h[2]; unsigned int u; };
union H4   { f16 h[4]; fv2 f; };

__device__ __forceinline__ float gelu_exact(float x) {
    return 0.5f * x * (1.0f + erff(x * 0.70710678118654752f));
}
__device__ __forceinline__ float softplus_f(float x) {
    return (x > 20.0f) ? x : log1pf(__expf(x));
}

// ---------------- P1: weight prep (f16 convert + A = -exp(A_logs)) ----------------
__global__ void p1_prep(const float* __restrict__ w1, const float* __restrict__ w2,
                        const float* __restrict__ alog,
                        f16* __restrict__ w1h, f16* __restrict__ w2h,
                        float* __restrict__ Aneg) {
    int t = blockIdx.x * blockDim.x + threadIdx.x;
    int stride = gridDim.x * blockDim.x;
    for (int i = t; i < 2*Dd*Cc; i += stride) w1h[i] = (f16)w1[i];
    for (int i = t; i < Cc*Dd;   i += stride) w2h[i] = (f16)w2[i];
    for (int i = t; i < Dd*Ns;   i += stride) Aneg[i] = -__expf(alog[i]);
}

// ---------------- P2: x (B,C,L) f32 -> xT (B,L,C) f16 ----------------
__global__ void p2_xT(const float* __restrict__ x, f16* __restrict__ xT) {
    int t = blockIdx.x * 256 + threadIdx.x;   // one thread per (b,l), B*L total
    int b = t >> 14;
    int l = t & (Ll - 1);
    const float* src = x + (size_t)b * Cc * Ll + l;
    f16* dst = xT + (size_t)t * Cc;
    #pragma unroll 8
    for (int c = 0; c < Cc; c += 2) {
        float v0 = src[(size_t)c * Ll];
        float v1 = src[(size_t)(c + 1) * Ll];
        H2 h; h.h[0] = (f16)v0; h.h[1] = (f16)v1;
        *(unsigned int*)(dst + c) = h.u;
    }
}

// ---------------- G1: in_proj GEMM (WMMA f16, 4 N-tiles per wave) ----------------
// out[m=d(0..255), n=l] = sum_c W1[d,c] * x[b,c,l]
// d<128 -> xin (B,D,L) f32 ; d>=128 -> zg = gelu(z) in (B,L,D) f32
__global__ void g1_inproj(const f16* __restrict__ w1h, const f16* __restrict__ xT,
                          float* __restrict__ xin, float* __restrict__ zg) {
    int lane = threadIdx.x & 31, wave = threadIdx.x >> 5;
    int grp = blockIdx.x * 8 + wave;        // Bn * 16 * 256 groups (4 l-tiles each)
    int lg = grp & 255;                     // 256 l-groups of 64 columns
    int dt = (grp >> 8) & 15;
    int b  = grp >> 12;
    int lane_lo = lane & 15, lane_hi = lane >> 4;
    int m0 = dt * 16, l0 = lg * 64;

    const f16* Abase = w1h + (size_t)(m0 + lane_lo) * Cc;              // row-major weights
    const f16* Bbase = xT  + ((size_t)b * Ll + l0 + lane_lo) * Cc;     // contiguous K per column

    v8f acc[4] = {{}, {}, {}, {}};
    #pragma unroll
    for (int kc = 0; kc < Cc; kc += 32) {
        Frag a;
        const f16* ap = Abase + kc + lane_hi * 8;
        a.f4[0] = *(const fv4*)(ap);
        a.f4[1] = *(const fv4*)(ap + 16);
        const f16* bp = Bbase + kc + lane_hi * 8;
        #pragma unroll
        for (int nt = 0; nt < 4; nt++) {
            Frag bf;
            const f16* bpn = bp + (size_t)nt * 16 * Cc;   // next 16 columns
            bf.f4[0] = *(const fv4*)(bpn);
            bf.f4[1] = *(const fv4*)(bpn + 16);
            acc[nt] = __builtin_amdgcn_wmma_f32_16x16x32_f16(false, a.v, false, bf.v,
                                                             (short)0, acc[nt], false, false);
        }
    }
    if (m0 < Dd) {
        float* dst = xin + (size_t)b * Dd * Ll;
        #pragma unroll
        for (int nt = 0; nt < 4; nt++) {
            int l = l0 + nt * 16 + lane_lo;
            #pragma unroll
            for (int i = 0; i < 8; i++) {
                int d = m0 + i + lane_hi * 8;
                dst[(size_t)d * Ll + l] = acc[nt][i];
            }
        }
    } else {
        #pragma unroll
        for (int nt = 0; nt < 4; nt++) {
            int l = l0 + nt * 16 + lane_lo;
            float* dst = zg + ((size_t)b * Ll + l) * Dd;
            #pragma unroll
            for (int i = 0; i < 8; i++) {
                int d = m0 - Dd + i + lane_hi * 8;
                dst[d] = gelu_exact(acc[nt][i]);
            }
        }
    }
}

// ---------------- C2: depthwise 3x3 conv + bias + gelu ----------------
__global__ void c2_dwconv_gelu(const float* __restrict__ xin, const float* __restrict__ cw,
                               const float* __restrict__ cb, float* __restrict__ xs) {
    int idx = blockIdx.x * 256 + threadIdx.x;   // B*D*L
    int w = idx & (Ww - 1);
    int h = (idx >> 7) & (Hh - 1);
    int dl = idx >> 14;                          // b*128 + d
    int d = dl & 127;
    const float* base = xin + (size_t)dl * Ll;
    float acc = cb[d];
    #pragma unroll
    for (int i = -1; i <= 1; i++) {
        int hh = h + i;
        if (hh < 0 || hh >= Hh) continue;
        #pragma unroll
        for (int j = -1; j <= 1; j++) {
            int ww2 = w + j;
            if (ww2 < 0 || ww2 >= Ww) continue;
            acc += base[hh * Ww + ww2] * cw[d * 9 + (i + 1) * 3 + (j + 1)];
        }
    }
    xs[idx] = gelu_exact(acc);
}

// ---------------- X3: x_proj (20 x 128, VALU) ----------------
__global__ void x3_xproj(const float* __restrict__ xs, const float* __restrict__ xw,
                         float* __restrict__ xdbl) {
    __shared__ float wsm[DBLc * Dd];
    for (int i = threadIdx.x; i < DBLc * Dd; i += 256) wsm[i] = xw[i];
    __syncthreads();
    int t = blockIdx.x * 256 + threadIdx.x;      // B*L threads
    int b = t >> 14;
    int l = t & (Ll - 1);
    float acc[DBLc];
    #pragma unroll
    for (int cc = 0; cc < DBLc; cc++) acc[cc] = 0.f;
    const float* src = xs + (size_t)b * Dd * Ll + l;
    for (int d = 0; d < Dd; d++) {
        float v = src[(size_t)d * Ll];
        #pragma unroll
        for (int cc = 0; cc < DBLc; cc++) acc[cc] += v * wsm[cc * Dd + d];
    }
    float* dst = xdbl + (size_t)b * DBLc * Ll + l;
    #pragma unroll
    for (int cc = 0; cc < DBLc; cc++) dst[(size_t)cc * Ll] = acc[cc];
}

// ---------------- X4: depthwise conv1d(k=7) + bias, dt_proj + softplus ----------------
__global__ void x4_conv1d_dt(const float* __restrict__ xdbl, const float* __restrict__ xcw,
                             const float* __restrict__ xcb, const float* __restrict__ dtw,
                             const float* __restrict__ dtb, float* __restrict__ BsCs,
                             float* __restrict__ delta) {
    __shared__ float s_dtw[Dd * Rr];
    __shared__ float s_dtb[Dd];
    __shared__ float s_cw[DBLc * 7];
    __shared__ float s_cb[DBLc];
    for (int i = threadIdx.x; i < Dd * Rr; i += 256) s_dtw[i] = dtw[i];
    for (int i = threadIdx.x; i < Dd;     i += 256) s_dtb[i] = dtb[i];
    for (int i = threadIdx.x; i < DBLc*7; i += 256) s_cw[i]  = xcw[i];
    for (int i = threadIdx.x; i < DBLc;   i += 256) s_cb[i]  = xcb[i];
    __syncthreads();
    int t = blockIdx.x * 256 + threadIdx.x;      // B*L threads
    int b = t >> 14;
    int l = t & (Ll - 1);
    float ch[DBLc];
    const float* src = xdbl + (size_t)b * DBLc * Ll;
    #pragma unroll
    for (int cc = 0; cc < DBLc; cc++) {
        float a = s_cb[cc];
        #pragma unroll
        for (int k = 0; k < 7; k++) {
            int ll = l + k - 3;
            if (ll >= 0 && ll < Ll) a += src[(size_t)cc * Ll + ll] * s_cw[cc * 7 + k];
        }
        ch[cc] = a;
    }
    float* bs = BsCs + (size_t)b * Ns * Ll + l;
    float* cs = BsCs + (size_t)Bn * Ns * Ll + (size_t)b * Ns * Ll + l;
    #pragma unroll
    for (int n = 0; n < Ns; n++) {
        bs[(size_t)n * Ll] = ch[Rr + n];
        cs[(size_t)n * Ll] = ch[Rr + Ns + n];
    }
    float* dst = delta + (size_t)b * Dd * Ll + l;
    for (int d = 0; d < Dd; d++) {
        float v = s_dtb[d];
        #pragma unroll
        for (int r = 0; r < Rr; r++) v += ch[r] * s_dtw[d * Rr + r];
        dst[(size_t)d * Ll] = softplus_f(v);
    }
}

// ---------------- S5a: per-chunk scan from h=0 -> (h_end, prod dA) ----------------
__global__ void s5a_chunk(const float* __restrict__ delta, const float* __restrict__ xs,
                          const float* __restrict__ BsCs, const float* __restrict__ Aneg,
                          float* __restrict__ hc, float* __restrict__ pc) {
    int t = blockIdx.x * 256 + threadIdx.x;  // 262144: (b,d,chunk) groups x 8 states
    int s = t & 7;
    int g = t >> 3;
    int chunk = g & (NCH - 1);
    int d = (g >> 6) & 127;
    int b = g >> 13;
    float A = Aneg[d * Ns + s];
    size_t rowbd = (size_t)(b * Dd + d) * Ll + chunk * CHL;
    const fv4* dp = (const fv4*)(delta + rowbd);
    const fv4* up = (const fv4*)(xs + rowbd);
    const fv4* bp = (const fv4*)(BsCs + (size_t)(b * Ns + s) * Ll + chunk * CHL);
    float h = 0.f, prod = 1.f;
    for (int i = 0; i < CHL / 4; i++) {
        __builtin_prefetch(dp + i + 8, 0, 0);
        __builtin_prefetch(up + i + 8, 0, 0);
        fv4 dl4 = dp[i], u4 = up[i], b4 = bp[i];
        #pragma unroll
        for (int k = 0; k < 4; k++) {
            float dl = dl4[k];
            float dA = __expf(dl * A);
            prod *= dA;
            h = dA * h + dl * u4[k] * b4[k];
        }
    }
    hc[t] = h;
    pc[t] = prod;
}

// ---------------- S5b: sequential chunk-prefix (tiny) ----------------
__global__ void s5b_seq(const float* __restrict__ hc, const float* __restrict__ pc,
                        float* __restrict__ hstart) {
    int t = blockIdx.x * 256 + threadIdx.x;  // 4096: (b,d,s)
    int s = t & 7;
    int d = (t >> 3) & 127;
    int b = t >> 10;
    size_t base = ((size_t)(b * Dd + d) * NCH) * 8 + s;
    float h = 0.f;
    for (int c = 0; c < NCH; c++) {
        size_t idx = base + (size_t)c * 8;
        hstart[idx] = h;
        h = pc[idx] * h + hc[idx];
    }
}

// ---------------- S5c: replay with correct h_start, emit y (B,L,D) ----------------
__global__ void s5c_scan(const float* __restrict__ delta, const float* __restrict__ xs,
                         const float* __restrict__ BsCs, const float* __restrict__ Aneg,
                         const float* __restrict__ hstart, const float* __restrict__ Dsv,
                         float* __restrict__ y) {
    int t = blockIdx.x * 256 + threadIdx.x;
    int s = t & 7;
    int g = t >> 3;
    int chunk = g & (NCH - 1);
    int d = (g >> 6) & 127;
    int b = g >> 13;
    float A = Aneg[d * Ns + s];
    float Dval = Dsv[d];
    size_t rowbd = (size_t)(b * Dd + d) * Ll + chunk * CHL;
    const fv4* dp = (const fv4*)(delta + rowbd);
    const fv4* up = (const fv4*)(xs + rowbd);
    const fv4* bp = (const fv4*)(BsCs + (size_t)(b * Ns + s) * Ll + chunk * CHL);
    const fv4* cp = (const fv4*)(BsCs + (size_t)Bn * Ns * Ll +
                                 (size_t)(b * Ns + s) * Ll + chunk * CHL);
    float h = hstart[t];
    float* ybase = y + ((size_t)b * Ll + chunk * CHL) * Dd + d;
    for (int i = 0; i < CHL / 4; i++) {
        __builtin_prefetch(dp + i + 8, 0, 0);
        __builtin_prefetch(up + i + 8, 0, 0);
        fv4 dl4 = dp[i], u4 = up[i], b4 = bp[i], c4 = cp[i];
        #pragma unroll
        for (int k = 0; k < 4; k++) {
            float dl = dl4[k];
            float dA = __expf(dl * A);
            h = dA * h + dl * u4[k] * b4[k];
            float p = h * c4[k];
            p += __shfl_xor(p, 1, 32);
            p += __shfl_xor(p, 2, 32);
            p += __shfl_xor(p, 4, 32);
            if (s == 0) ybase[(size_t)(i * 4 + k) * Dd] = p + Dval * u4[k];
        }
    }
}

// ---------------- L6: LayerNorm(D=128) * gelu(z) -> f16 (B,L,D) ----------------
__global__ void l6_ln_gate(const float* __restrict__ y, const float* __restrict__ zg,
                           const float* __restrict__ g, const float* __restrict__ bta,
                           f16* __restrict__ ynh) {
    int lane = threadIdx.x & 31;
    int row = blockIdx.x * 8 + (threadIdx.x >> 5);   // B*L rows, one per wave
    const fv4* yp = (const fv4*)(y + (size_t)row * Dd);
    fv4 v = yp[lane];
    float sum = v[0] + v[1] + v[2] + v[3];
    float sq  = v[0]*v[0] + v[1]*v[1] + v[2]*v[2] + v[3]*v[3];
    #pragma unroll
    for (int m = 1; m < 32; m <<= 1) {
        sum += __shfl_xor(sum, m, 32);
        sq  += __shfl_xor(sq,  m, 32);
    }
    float mean = sum * (1.f / Dd);
    float var  = sq * (1.f / Dd) - mean * mean;
    float rstd = rsqrtf(var + 1e-5f);
    const fv4* zp = (const fv4*)(zg + (size_t)row * Dd);
    fv4 z4 = zp[lane];
    H4 o;
    #pragma unroll
    for (int i = 0; i < 4; i++) {
        int dd = lane * 4 + i;
        float yn = (v[i] - mean) * rstd * g[dd] + bta[dd];
        o.h[i] = (f16)(yn * z4[i]);
    }
    *(fv2*)(ynh + (size_t)row * Dd + lane * 4) = o.f;
}

// ---------------- G7: out_proj GEMM (WMMA f16, 4 N-tiles per wave) ----------------
__global__ void g7_outproj(const f16* __restrict__ w2h, const f16* __restrict__ ynh,
                           float* __restrict__ out) {
    int lane = threadIdx.x & 31, wave = threadIdx.x >> 5;
    int grp = blockIdx.x * 8 + wave;     // Bn * 4 * 256 groups (4 l-tiles each)
    int lg = grp & 255;
    int ct = (grp >> 8) & 3;
    int b  = grp >> 10;
    int lane_lo = lane & 15, lane_hi = lane >> 4;
    int m0 = ct * 16, l0 = lg * 64;

    const f16* Abase = w2h + (size_t)(m0 + lane_lo) * Dd;
    const f16* Bbase = ynh + ((size_t)b * Ll + l0 + lane_lo) * Dd;

    v8f acc[4] = {{}, {}, {}, {}};
    #pragma unroll
    for (int kc = 0; kc < Dd; kc += 32) {
        Frag a;
        const f16* ap = Abase + kc + lane_hi * 8;
        a.f4[0] = *(const fv4*)(ap);
        a.f4[1] = *(const fv4*)(ap + 16);
        const f16* bp = Bbase + kc + lane_hi * 8;
        #pragma unroll
        for (int nt = 0; nt < 4; nt++) {
            Frag bf;
            const f16* bpn = bp + (size_t)nt * 16 * Dd;
            bf.f4[0] = *(const fv4*)(bpn);
            bf.f4[1] = *(const fv4*)(bpn + 16);
            acc[nt] = __builtin_amdgcn_wmma_f32_16x16x32_f16(false, a.v, false, bf.v,
                                                             (short)0, acc[nt], false, false);
        }
    }
    float* dstb = out + (size_t)b * Cc * Ll;
    #pragma unroll
    for (int nt = 0; nt < 4; nt++) {
        int l = l0 + nt * 16 + lane_lo;
        #pragma unroll
        for (int i = 0; i < 8; i++) {
            int c = m0 + i + lane_hi * 8;
            dstb[(size_t)c * Ll + l] = acc[nt][i];
        }
    }
}

// ---------------- host launch ----------------
extern "C" void kernel_launch(void* const* d_in, const int* in_sizes, int n_in,
                              void* d_out, int out_size, void* d_ws, size_t ws_size,
                              hipStream_t stream) {
    const float* x    = (const float*)d_in[0];
    const float* w1   = (const float*)d_in[1];
    const float* cw   = (const float*)d_in[2];
    const float* cb   = (const float*)d_in[3];
    const float* xpw  = (const float*)d_in[4];
    const float* xcw  = (const float*)d_in[5];
    const float* xcb  = (const float*)d_in[6];
    const float* dtw  = (const float*)d_in[7];
    const float* dtb  = (const float*)d_in[8];
    const float* alog = (const float*)d_in[9];
    const float* Dsv  = (const float*)d_in[10];
    const float* lng  = (const float*)d_in[11];
    const float* lnb  = (const float*)d_in[12];
    const float* w2   = (const float*)d_in[13];
    float* out = (float*)d_out;

    char* ws = (char*)d_ws;
    size_t off = 0;
    auto alloc = [&](size_t bytes) -> char* {
        char* p = ws + off;
        off = (off + bytes + 511) & ~(size_t)511;
        return p;
    };

    f16*   w1h    = (f16*)  alloc((size_t)2 * Dd * Cc * 2);
    f16*   w2h    = (f16*)  alloc((size_t)Cc * Dd * 2);
    float* Aneg   = (float*)alloc((size_t)Dd * Ns * 4);
    f16*   xT     = (f16*)  alloc((size_t)Bn * Ll * Cc * 2);
    float* buf1   = (float*)alloc((size_t)Bn * Dd * Ll * 4);   // xin, later reused as y
    float* zg     = (float*)alloc((size_t)Bn * Ll * Dd * 4);
    float* xs     = (float*)alloc((size_t)Bn * Dd * Ll * 4);
    float* xdbl   = (float*)alloc((size_t)Bn * DBLc * Ll * 4);
    float* delta  = (float*)alloc((size_t)Bn * Dd * Ll * 4);
    float* BsCs   = (float*)alloc((size_t)Bn * 2 * Ns * Ll * 4);
    f16*   ynh    = (f16*)  alloc((size_t)Bn * Ll * Dd * 2);
    float* hc     = (float*)alloc((size_t)Bn * Dd * NCH * Ns * 4);
    float* pc     = (float*)alloc((size_t)Bn * Dd * NCH * Ns * 4);
    float* hstart = (float*)alloc((size_t)Bn * Dd * NCH * Ns * 4);

    float* xin = buf1;
    float* yb  = buf1;   // xin is dead after c2; reuse for scan output y

    p1_prep<<<64, 256, 0, stream>>>(w1, w2, alog, w1h, w2h, Aneg);
    p2_xT<<<(Bn * Ll) / 256, 256, 0, stream>>>(x, xT);
    g1_inproj<<<(Bn * 16 * (Ll / 64)) / 8, 256, 0, stream>>>(w1h, xT, xin, zg);
    c2_dwconv_gelu<<<(Bn * Dd * Ll) / 256, 256, 0, stream>>>(xin, cw, cb, xs);
    x3_xproj<<<(Bn * Ll) / 256, 256, 0, stream>>>(xs, xpw, xdbl);
    x4_conv1d_dt<<<(Bn * Ll) / 256, 256, 0, stream>>>(xdbl, xcw, xcb, dtw, dtb, BsCs, delta);
    s5a_chunk<<<(Bn * Dd * NCH * Ns) / 256, 256, 0, stream>>>(delta, xs, BsCs, Aneg, hc, pc);
    s5b_seq<<<(Bn * Dd * Ns) / 256, 256, 0, stream>>>(hc, pc, hstart);
    s5c_scan<<<(Bn * Dd * NCH * Ns) / 256, 256, 0, stream>>>(delta, xs, BsCs, Aneg, hstart, Dsv, yb);
    l6_ln_gate<<<(Bn * Ll) / 8, 256, 0, stream>>>(yb, zg, lng, lnb, ynh);
    g7_outproj<<<(Bn * 4 * (Ll / 64)) / 8, 256, 0, stream>>>(w2h, ynh, out);

    (void)in_sizes; (void)n_in; (void)out_size; (void)ws_size;
}
